// FusedAttention_31172872634644
// MI455X (gfx1250) — compile-verified, matrix operand order checked by reading, not run on hardware
//
#include <hip/hip_runtime.h>

// ---------------------------------------------------------------------------
// Fused attention block for MI455X (gfx1250, wave32, WMMA bf16 path)
// B=2, S=1024, HIDDEN=4096, HQ=32, HKV=8, D=128
// Pipeline: bf16 convert -> Q/KV proj (WMMA GEMM, async-LDS double buffered)
//           -> RoPE+scatter -> causal GQA flash attention (WMMA) -> O proj.
// ---------------------------------------------------------------------------

typedef __attribute__((ext_vector_type(16))) __bf16 v16bf;
typedef __attribute__((ext_vector_type(8)))  float  v8f;

#define B_     2
#define S_     1024
#define HID_   4096
#define HQ_    32
#define HKV_   8
#define D_     128
#define TOKS_  (B_ * S_)          // 2048

// ------------------------- async global->LDS path --------------------------
#if defined(__has_builtin)
#if __has_builtin(__builtin_amdgcn_global_load_async_to_lds_b128)
#define HAVE_ASYNC_LDS 1
#endif
#endif
#ifndef HAVE_ASYNC_LDS
#define HAVE_ASYNC_LDS 0
#endif

// builtin signature: (int4 AS1* gsrc, int4 AS3* lds, imm offset, imm cpol)
typedef int v4i_ __attribute__((vector_size(16)));
typedef __attribute__((address_space(1))) v4i_ gv4i_t;
typedef __attribute__((address_space(3))) v4i_ lv4i_t;

// copy 16B (8 bf16) from global to LDS
__device__ __forceinline__ void cp16_to_lds(unsigned short* lds_dst,
                                            const unsigned short* gsrc) {
#if HAVE_ASYNC_LDS
  __builtin_amdgcn_global_load_async_to_lds_b128(
      (gv4i_t*)(v4i_*)(void*)const_cast<unsigned short*>(gsrc),
      (lv4i_t*)(v4i_*)(void*)lds_dst, 0, 0);
#else
  *reinterpret_cast<uint4*>(lds_dst) = *reinterpret_cast<const uint4*>(gsrc);
#endif
}

__device__ __forceinline__ void async_wait_all() {
#if HAVE_ASYNC_LDS
#if __has_builtin(__builtin_amdgcn_s_wait_asynccnt)
  __builtin_amdgcn_s_wait_asynccnt(0);
#else
  asm volatile("s_wait_asynccnt 0" ::: "memory");
#endif
#endif
}

// ----------------------------- small helpers -------------------------------
__device__ __forceinline__ unsigned short f32_to_bf16(float f) {
  unsigned int u = __float_as_uint(f);
  unsigned int r = (u + 0x7FFFu + ((u >> 16) & 1u)) >> 16;   // RNE
  return (unsigned short)r;
}

__device__ __forceinline__ v16bf load_frag16(const unsigned short* lo,
                                             const unsigned short* hi) {
  union { uint4 u[2]; v16bf v; } f;
  f.u[0] = *reinterpret_cast<const uint4*>(lo);
  f.u[1] = *reinterpret_cast<const uint4*>(hi);
  return f.v;
}

// ---------------------------------------------------------------------------
// f32 -> bf16, 4 elements per thread (float4 in, 8B out)
// ---------------------------------------------------------------------------
__global__ void to_bf16_kernel(const float* __restrict__ in,
                               unsigned short* __restrict__ out, int n4) {
  int i = blockIdx.x * blockDim.x + threadIdx.x;
  if (i >= n4) return;
  float4 v = reinterpret_cast<const float4*>(in)[i];
  union { unsigned short s[4]; uint2 u; } p;
  p.s[0] = f32_to_bf16(v.x);
  p.s[1] = f32_to_bf16(v.y);
  p.s[2] = f32_to_bf16(v.z);
  p.s[3] = f32_to_bf16(v.w);
  reinterpret_cast<uint2*>(out)[i] = p.u;
}

// ---------------------------------------------------------------------------
// GEMM: C[M,N] = A[M,K] * W[N,K]^T  (A,W bf16 row-major, C f32 row-major)
// 128x128x32 tiles; 256 threads = 8 waves (2x4); 4x2 WMMA tiles per wave.
// Double-buffered LDS, async global->LDS staging.
// ---------------------------------------------------------------------------
#define GBM 128
#define GBN 128
#define GBK 32
#define GLD 40     // padded LDS row stride (elements): conflict-free

__global__ __launch_bounds__(256)
void gemm_bf16_nt(const unsigned short* __restrict__ A,
                  const unsigned short* __restrict__ W,
                  float* __restrict__ C, int M, int N, int K) {
  __shared__ unsigned short As[2][GBM * GLD];
  __shared__ unsigned short Bs[2][GBN * GLD];

  const int tid  = threadIdx.x;
  const int lane = tid & 31;
  const int wave = tid >> 5;
  const int wm   = wave & 1;         // 0..1 -> 64-row slab
  const int wn   = wave >> 1;        // 0..3 -> 32-col slab
  const int half = lane >> 4;
  const int l16  = lane & 15;
  const int kb8  = half * 8;

  const int m0 = blockIdx.y * GBM;
  const int n0 = blockIdx.x * GBN;

  // this thread's two staging chunks: chunk c -> row = c>>2, col = (c&3)*8
  const int r0 = tid >> 2,          c0 = (tid & 3) << 3;
  const int r1 = (tid + 256) >> 2,  c1 = ((tid + 256) & 3) << 3;
  const unsigned short* gA0 = A + (size_t)(m0 + r0) * K + c0;
  const unsigned short* gA1 = A + (size_t)(m0 + r1) * K + c1;
  const unsigned short* gB0 = W + (size_t)(n0 + r0) * K + c0;
  const unsigned short* gB1 = W + (size_t)(n0 + r1) * K + c1;

  v8f acc[4][2];
#pragma unroll
  for (int i = 0; i < 4; ++i)
#pragma unroll
    for (int j = 0; j < 2; ++j) acc[i][j] = (v8f){0, 0, 0, 0, 0, 0, 0, 0};

  // prologue: stage tile 0 into buffer 0
  cp16_to_lds(&As[0][r0 * GLD + c0], gA0);
  cp16_to_lds(&As[0][r1 * GLD + c1], gA1);
  cp16_to_lds(&Bs[0][r0 * GLD + c0], gB0);
  cp16_to_lds(&Bs[0][r1 * GLD + c1], gB1);
  async_wait_all();
  __syncthreads();

  int buf = 0;
  for (int kt = 0; kt < K; kt += GBK) {
    // stage next K tile into the other buffer (async; no wait yet)
    if (kt + GBK < K) {
      const int ko = kt + GBK;
      cp16_to_lds(&As[buf ^ 1][r0 * GLD + c0], gA0 + ko);
      cp16_to_lds(&As[buf ^ 1][r1 * GLD + c1], gA1 + ko);
      cp16_to_lds(&Bs[buf ^ 1][r0 * GLD + c0], gB0 + ko);
      cp16_to_lds(&Bs[buf ^ 1][r1 * GLD + c1], gB1 + ko);
    }
    if (kt + 2 * GBK < K) {   // speculative prefetch of the tile after next
      __builtin_prefetch(gA0 + kt + 2 * GBK, 0, 3);
      __builtin_prefetch(gB0 + kt + 2 * GBK, 0, 3);
    }

    // compute on current buffer
    v16bf aF[4], bF[2];
#pragma unroll
    for (int mt = 0; mt < 4; ++mt) {
      const unsigned short* p = &As[buf][(wm * 64 + mt * 16 + l16) * GLD];
      aF[mt] = load_frag16(p + kb8, p + 16 + kb8);
    }
#pragma unroll
    for (int nt = 0; nt < 2; ++nt) {
      const unsigned short* p = &Bs[buf][(wn * 32 + nt * 16 + l16) * GLD];
      bF[nt] = load_frag16(p + kb8, p + 16 + kb8);
    }
#pragma unroll
    for (int mt = 0; mt < 4; ++mt)
#pragma unroll
      for (int nt = 0; nt < 2; ++nt)
        acc[mt][nt] = __builtin_amdgcn_wmma_f32_16x16x32_bf16(
            false, aF[mt], false, bF[nt], (short)0, acc[mt][nt], false, false);

    async_wait_all();     // next tile fully landed in LDS
    __syncthreads();      // all waves done reading current buffer
    buf ^= 1;
  }

  // epilogue: C layout (VGPR r -> M = r + 8*half; N = lane%16)
#pragma unroll
  for (int mt = 0; mt < 4; ++mt)
#pragma unroll
    for (int nt = 0; nt < 2; ++nt)
#pragma unroll
      for (int r = 0; r < 8; ++r) {
        int row = m0 + wm * 64 + mt * 16 + r + 8 * half;
        int col = n0 + wn * 32 + nt * 16 + l16;
        C[(size_t)row * N + col] = acc[mt][nt][r];
      }
}

// ---------------------------------------------------------------------------
// RoPE on Q (half-rotation layout), f32 -> bf16, layout [tok, HQ, D]
// ---------------------------------------------------------------------------
__global__ void rope_q_kernel(const float* __restrict__ xq,
                              const float* __restrict__ cosT,
                              const float* __restrict__ sinT,
                              unsigned short* __restrict__ Qb) {
  int idx = blockIdx.x * blockDim.x + threadIdx.x;   // TOKS*HQ*D
  if (idx >= TOKS_ * HQ_ * D_) return;
  int d    = idx & (D_ - 1);
  int tok  = idx >> 12;                 // / (HQ*D)
  int pos  = tok & (S_ - 1);
  float v   = xq[idx];
  float rot = (d < 64) ? -xq[idx + 64] : xq[idx - 64];
  float o   = v * cosT[pos * D_ + d] + rot * sinT[pos * D_ + d];
  Qb[idx]   = f32_to_bf16(o);
}

// ---------------------------------------------------------------------------
// RoPE on K + KV-cache scatter via cur_select_index; V copied. bf16 out,
// layout [tok, HKV, D].
// ---------------------------------------------------------------------------
__global__ void rope_kv_kernel(const float* __restrict__ xkv,
                               const float* __restrict__ cosT,
                               const float* __restrict__ sinT,
                               const int* __restrict__ sel,
                               unsigned short* __restrict__ Kb,
                               unsigned short* __restrict__ Vb) {
  int idx = blockIdx.x * blockDim.x + threadIdx.x;   // TOKS*HKV*D
  if (idx >= TOKS_ * HKV_ * D_) return;
  int d   = idx & (D_ - 1);
  int hk  = (idx >> 7) & (HKV_ - 1);
  int tok = idx >> 10;
  int pos = tok & (S_ - 1);
  const float* row = xkv + (size_t)tok * (2 * HKV_ * D_);
  float kv  = row[hk * D_ + d];
  float rot = (d < 64) ? -row[hk * D_ + d + 64] : row[hk * D_ + d - 64];
  float kr  = kv * cosT[pos * D_ + d] + rot * sinT[pos * D_ + d];
  float vv  = row[HKV_ * D_ + hk * D_ + d];
  int dst   = sel[tok];
  size_t o  = (size_t)dst * (HKV_ * D_) + hk * D_ + d;
  Kb[o] = f32_to_bf16(kr);
  Vb[o] = f32_to_bf16(vv);
}

// ---------------------------------------------------------------------------
// Flash attention (causal, GQA 4:1). Grid = B*HQ*(S/64), block = 128 (4 waves)
// Each wave owns 16 q rows; K/V tiles of 32 keys staged in LDS (K via async).
// ---------------------------------------------------------------------------
#define KLD 136    // Ks row stride (32 keys x 128 dims, padded)
#define VLD 40     // Vt row stride (128 dims x 32 keys, padded)

__global__ __launch_bounds__(128)
void flash_attn_kernel(const unsigned short* __restrict__ Qb,   // [TOKS,HQ,D]
                       const unsigned short* __restrict__ Kb,   // [TOKS,HKV,D]
                       const unsigned short* __restrict__ Vb,   // [TOKS,HKV,D]
                       unsigned short* __restrict__ Ob) {       // [TOKS,HQ,D]
  __shared__ unsigned short Ks[32 * KLD];
  __shared__ unsigned short Vt[128 * VLD];
  __shared__ unsigned short Ps[4][16 * VLD];

  const int tid  = threadIdx.x;
  const int lane = tid & 31;
  const int wave = tid >> 5;
  const int half = lane >> 4;
  const int l16  = lane & 15;
  const int kb8  = half * 8;

  const int bid = blockIdx.x;
  const int qt  = bid & 15;            // S/64 = 16 q tiles
  const int h   = (bid >> 4) & (HQ_ - 1);
  const int b   = bid >> 9;
  const int hk  = h >> 2;              // GQA group of 4
  const int q0  = qt * 64;
  const int qw0 = q0 + wave * 16;      // this wave's first q row in sequence

  // Q fragments (16 rows x 128 dims = 4 K-chunks), held in registers
  v16bf qF[4];
  {
    const unsigned short* qp =
        Qb + ((size_t)(b * S_ + qw0 + l16) * HQ_ + h) * D_;
#pragma unroll
    for (int kc = 0; kc < 4; ++kc)
      qF[kc] = load_frag16(qp + kc * 32 + kb8, qp + kc * 32 + 16 + kb8);
  }

  float mrow[8], lrow[8];
  v8f oacc[8];
#pragma unroll
  for (int r = 0; r < 8; ++r) { mrow[r] = -3.0e38f; lrow[r] = 0.0f; }
#pragma unroll
  for (int nt = 0; nt < 8; ++nt) oacc[nt] = (v8f){0, 0, 0, 0, 0, 0, 0, 0};

  const float scale = 0.088388347648318447f;   // 1/sqrt(128)

  for (int k0 = 0; k0 < q0 + 64; k0 += 32) {
    // stage K tile (async to LDS, row-major) and V tile (reg transpose)
#pragma unroll
    for (int i = 0; i < 4; ++i) {
      int c   = tid + i * 128;           // 0..511
      int key = c >> 4;
      int col = (c & 15) << 3;
      size_t gbase = ((size_t)(b * S_ + k0 + key) * HKV_ + hk) * D_ + col;
      cp16_to_lds(&Ks[key * KLD + col], Kb + gbase);
      uint4 vv4 = *reinterpret_cast<const uint4*>(Vb + gbase);
      const unsigned short* vs = reinterpret_cast<const unsigned short*>(&vv4);
#pragma unroll
      for (int j = 0; j < 8; ++j) Vt[(col + j) * VLD + key] = vs[j];
    }
    async_wait_all();
    __syncthreads();

    // S = Q * K^T : two 16x16 score tiles, accumulate over 4 K-chunks
    v8f sacc[2];
    sacc[0] = (v8f){0, 0, 0, 0, 0, 0, 0, 0};
    sacc[1] = (v8f){0, 0, 0, 0, 0, 0, 0, 0};
#pragma unroll
    for (int kc = 0; kc < 4; ++kc)
#pragma unroll
      for (int nt = 0; nt < 2; ++nt) {
        const unsigned short* p = &Ks[(nt * 16 + l16) * KLD + kc * 32];
        v16bf kF = load_frag16(p + kb8, p + 16 + kb8);
        sacc[nt] = __builtin_amdgcn_wmma_f32_16x16x32_bf16(
            false, qF[kc], false, kF, (short)0, sacc[nt], false, false);
      }

    const bool maskTile = (k0 + 31 > qw0);
    float rscale[8];
#pragma unroll
    for (int r = 0; r < 8; ++r) {
      float s0 = sacc[0][r] * scale;
      float s1 = sacc[1][r] * scale;
      if (maskTile) {
        int qg = qw0 + r + 8 * half;
        if (k0 + l16 > qg)      s0 = -1.0e30f;
        if (k0 + 16 + l16 > qg) s1 = -1.0e30f;
      }
      float mx = fmaxf(s0, s1);
#pragma unroll
      for (int off = 1; off < 16; off <<= 1)
        mx = fmaxf(mx, __shfl_xor(mx, off, 32));
      float mnew = fmaxf(mrow[r], mx);
      float p0 = __expf(s0 - mnew);
      float p1 = __expf(s1 - mnew);
      rscale[r] = __expf(mrow[r] - mnew);
      mrow[r] = mnew;
      float ps = p0 + p1;
#pragma unroll
      for (int off = 1; off < 16; off <<= 1)
        ps += __shfl_xor(ps, off, 32);
      lrow[r] = lrow[r] * rscale[r] + ps;
      // write P (bf16) to per-wave LDS scratch for A-layout reload
      int prow = r + 8 * half;
      Ps[wave][prow * VLD + l16]      = f32_to_bf16(p0);
      Ps[wave][prow * VLD + 16 + l16] = f32_to_bf16(p1);
    }

    // rescale running output
#pragma unroll
    for (int nt = 0; nt < 8; ++nt)
#pragma unroll
      for (int r = 0; r < 8; ++r) oacc[nt][r] *= rscale[r];

    // O += P * V  (A = P 16x32, B = V^T fragments from transposed LDS)
    const unsigned short* pp = &Ps[wave][l16 * VLD];
    v16bf pF = load_frag16(pp + kb8, pp + 16 + kb8);
#pragma unroll
    for (int nt = 0; nt < 8; ++nt) {
      const unsigned short* vp = &Vt[(nt * 16 + l16) * VLD];
      v16bf vF = load_frag16(vp + kb8, vp + 16 + kb8);
      oacc[nt] = __builtin_amdgcn_wmma_f32_16x16x32_bf16(
          false, pF, false, vF, (short)0, oacc[nt], false, false);
    }
    __syncthreads();
  }

  // finalize: divide by l, store bf16 [tok, h, dim]
#pragma unroll
  for (int r = 0; r < 8; ++r) {
    float invl = 1.0f / lrow[r];
    size_t tok = (size_t)(b * S_ + qw0 + r + 8 * half);
#pragma unroll
    for (int nt = 0; nt < 8; ++nt)
      Ob[(tok * HQ_ + h) * D_ + nt * 16 + l16] =
          f32_to_bf16(oacc[nt][r] * invl);
  }
}

// ---------------------------------------------------------------------------
// Host launch
// ---------------------------------------------------------------------------
extern "C" void kernel_launch(void* const* d_in, const int* in_sizes, int n_in,
                              void* d_out, int out_size, void* d_ws, size_t ws_size,
                              hipStream_t stream) {
  const float* x    = (const float*)d_in[0];   // [2,1024,4096]
  const float* q_w  = (const float*)d_in[1];   // [4096,4096]
  const float* kv_w = (const float*)d_in[2];   // [2048,4096]
  const float* o_w  = (const float*)d_in[3];   // [4096,4096]
  const float* cosT = (const float*)d_in[4];   // [1024,128]
  const float* sinT = (const float*)d_in[5];   // [1024,128]
  const int*   sel  = (const int*)d_in[6];     // [2048]
  float* out = (float*)d_out;                  // [2048,4096]

  char* ws = (char*)d_ws;
  unsigned short* x_bf    = (unsigned short*)ws; ws += (size_t)TOKS_ * HID_ * 2;
  unsigned short* qw_bf   = (unsigned short*)ws; ws += (size_t)HID_ * HID_ * 2;
  unsigned short* kvw_bf  = (unsigned short*)ws; ws += (size_t)2 * HKV_ * D_ * HID_ * 2;
  unsigned short* ow_bf   = (unsigned short*)ws; ws += (size_t)HID_ * HID_ * 2;
  float*          xq_f    = (float*)ws;          ws += (size_t)TOKS_ * HID_ * 4;
  float*          xkv_f   = (float*)ws;          ws += (size_t)TOKS_ * 2 * HKV_ * D_ * 4;
  unsigned short* q_bf    = (unsigned short*)ws; ws += (size_t)TOKS_ * HQ_ * D_ * 2;
  unsigned short* k_bf    = (unsigned short*)ws; ws += (size_t)TOKS_ * HKV_ * D_ * 2;
  unsigned short* v_bf    = (unsigned short*)ws; ws += (size_t)TOKS_ * HKV_ * D_ * 2;
  unsigned short* attn_bf = (unsigned short*)ws; ws += (size_t)TOKS_ * HQ_ * D_ * 2;

  const int nx  = TOKS_ * HID_;          // 8388608
  const int nqw = HID_ * HID_;           // 16777216
  const int nkv = 2 * HKV_ * D_ * HID_;  // 8388608

  // 1) f32 -> bf16 conversions (4 elems/thread)
  to_bf16_kernel<<<(nx / 4  + 255) / 256, 256, 0, stream>>>(x,    x_bf,   nx / 4);
  to_bf16_kernel<<<(nqw / 4 + 255) / 256, 256, 0, stream>>>(q_w,  qw_bf,  nqw / 4);
  to_bf16_kernel<<<(nkv / 4 + 255) / 256, 256, 0, stream>>>(kv_w, kvw_bf, nkv / 4);
  to_bf16_kernel<<<(nqw / 4 + 255) / 256, 256, 0, stream>>>(o_w,  ow_bf,  nqw / 4);

  // 2) projections: xq = x @ q_w^T ; xkv = x @ kv_w^T
  gemm_bf16_nt<<<dim3(HID_ / GBN, TOKS_ / GBM), 256, 0, stream>>>(
      x_bf, qw_bf, xq_f, TOKS_, HID_, HID_);
  gemm_bf16_nt<<<dim3((2 * HKV_ * D_) / GBN, TOKS_ / GBM), 256, 0, stream>>>(
      x_bf, kvw_bf, xkv_f, TOKS_, 2 * HKV_ * D_, HID_);

  // 3) RoPE + KV scatter
  rope_q_kernel<<<(TOKS_ * HQ_ * D_ + 255) / 256, 256, 0, stream>>>(
      xq_f, cosT, sinT, q_bf);
  rope_kv_kernel<<<(TOKS_ * HKV_ * D_ + 255) / 256, 256, 0, stream>>>(
      xkv_f, cosT, sinT, sel, k_bf, v_bf);

  // 4) causal GQA flash attention
  flash_attn_kernel<<<B_ * HQ_ * (S_ / 64), 128, 0, stream>>>(
      q_bf, k_bf, v_bf, attn_bf);

  // 5) output projection: out = attn @ o_w^T  (f32 result)
  gemm_bf16_nt<<<dim3(HID_ / GBN, TOKS_ / GBM), 256, 0, stream>>>(
      attn_bf, ow_bf, out, TOKS_, HID_, HID_);
}
